// SeedFormer_88038239634151
// MI455X (gfx1250) — compile-verified, matrix operand order checked by reading, not run on hardware
//
#include <hip/hip_runtime.h>
#include <math.h>
#include <stdint.h>

// Problem constants (match reference)
#define Bn    8
#define Npt   2048
#define KNN   20
#define UPF   2
#define CINC  256
#define DD    64
#define COUTC 128
#define PHH   64
#define AHH   256

typedef __attribute__((ext_vector_type(16))) _Float16 v16h;
typedef __attribute__((ext_vector_type(8)))  float    v8f;

// CDNA5 16-bit A/B fragment element -> K offset within a 32-wide K slab.
// Per ISA 7.12.2: lanes 0-15 hold K=0..7 (VGPR0-3) and K=16..23 (VGPR4-7),
// lanes 16-31 hold K=8..15 and K=24..31; each VGPR packs a K pair.
__device__ __forceinline__ int frag_k(int i, int half) {
  int j = i >> 1, p = i & 1;
  return ((j >> 2) << 4) + (half << 3) + ((j & 3) << 1) + p;
}

// A fragment from a row-major (O x C) fp32 weight matrix, converted to f16.
__device__ __forceinline__ v16h load_a_rowmajor(const float* __restrict__ W,
                                                int ld, int m0, int c0, int lane) {
  int m = m0 + (lane & 15);
  int half = lane >> 4;
  v16h a;
#pragma unroll
  for (int i = 0; i < 16; ++i)
    a[i] = (_Float16)W[(long)m * ld + c0 + frag_k(i, half)];
  return a;
}

// CDNA5 async copy: 16 bytes global -> LDS, tracked by ASYNCcnt.
__device__ __forceinline__ void async_copy_b128(uint32_t lds_off, const float* gaddr) {
  asm volatile("global_load_async_to_lds_b128 %0, %1, off"
               :: "v"(lds_off), "v"(gaddr) : "memory");
}
__device__ __forceinline__ void wait_async0() {
  asm volatile("s_wait_asynccnt 0x0" ::: "memory");
}

// ---------------------------------------------------------------------------
// kNN: one thread per (b,n), maintain running top-20 smallest squared dists.
// ---------------------------------------------------------------------------
__global__ __launch_bounds__(256) void knn_kernel(const float* __restrict__ pos,
                                                  int* __restrict__ idx) {
  int t = blockIdx.x * 256 + threadIdx.x;
  if (t >= Bn * Npt) return;
  int b = t / Npt, n = t % Npt;
  const float* px = pos + (long)b * 3 * Npt;
  float qx = px[n], qy = px[Npt + n], qz = px[2 * Npt + n];
  float bd[KNN]; int bi[KNN];
#pragma unroll
  for (int i = 0; i < KNN; ++i) { bd[i] = 3.4e38f; bi[i] = n; }
  float worst = 3.4e38f; int wp = 0;
  for (int m = 0; m < Npt; ++m) {
    float dx = qx - px[m], dy = qy - px[Npt + m], dz = qz - px[2 * Npt + m];
    float d = dx * dx + dy * dy + dz * dz;
    if (d < worst) {
      bd[wp] = d; bi[wp] = m;
      worst = bd[0]; wp = 0;
#pragma unroll
      for (int j = 1; j < KNN; ++j)
        if (bd[j] > worst) { worst = bd[j]; wp = j; }
    }
  }
  int* out = idx + (long)t * KNN;
#pragma unroll
  for (int i = 0; i < KNN; ++i) out[i] = bi[i];
}

// ---------------------------------------------------------------------------
// 4-wave WMMA GEMM, 64x16 output tile per block, B-slab staged in LDS via
// global_load_async_to_lds_b128 (ASYNCcnt) and shared by all 4 waves:
//   Y(b,O,Ntok) (+)= W(O,C) * X(b,C,Ntok) + bias  [, relu][, + resid repeated]
// Concat input supported: rows [0,Chalf) from X, rows [Chalf,C) from X2.
// Requires O % 64 == 0, C % 32 == 0, Ntok % 16 == 0.
// ---------------------------------------------------------------------------
__global__ __launch_bounds__(128) void gemm16(
    const float* __restrict__ X, const float* __restrict__ X2, int Chalf,
    const float* __restrict__ W, const float* __restrict__ bias,
    const float* __restrict__ resid, int rrep,
    float* __restrict__ Y, int O, int C, int Ntok, int relu, int acc) {
  __shared__ float sX[32][16];   // one K-slab of B: 32 rows x 16 tokens (fp32)

  const int tid  = threadIdx.x;
  const int w    = tid >> 5;
  const int lane = tid & 31;
  const int half = lane >> 4;
  const int ln   = lane & 15;
  const int n0   = blockIdx.x * 16;
  const int m0   = blockIdx.y * 64 + w * 16;
  const int b    = blockIdx.z;

  const float* Xb  = X + (long)b * Chalf * Ntok;
  const float* X2b = (Chalf < C) ? (X2 + (long)b * (C - Chalf) * Ntok) : X;
  const uint32_t ldsbase = (uint32_t)(size_t)&sX[0][0];
  const int srow = tid >> 2;        // 0..31: slab row this thread stages
  const int scg  = (tid & 3) * 4;   // column group (4 floats = 16B)

  v8f accv = {0.f, 0.f, 0.f, 0.f, 0.f, 0.f, 0.f, 0.f};
  for (int c0 = 0; c0 < C; c0 += 32) {
    // Stage B slab: 128 threads x b128 = 2KB, coalesced 64B rows.
    {
      int c = c0 + srow;
      const float* src = (c < Chalf) ? (Xb + (long)c * Ntok)
                                     : (X2b + (long)(c - Chalf) * Ntok);
      async_copy_b128(ldsbase + (uint32_t)((srow * 16 + scg) * sizeof(float)),
                      src + n0 + scg);
    }
    wait_async0();
    __syncthreads();

    if (c0 + 64 <= C)
      __builtin_prefetch(&W[(long)m0 * C + c0 + 64], 0, 1);  // global_prefetch_b8
    v16h a = load_a_rowmajor(W, C, m0, c0, lane);
    v16h bf;
#pragma unroll
    for (int i = 0; i < 16; ++i)
      bf[i] = (_Float16)sX[frag_k(i, half)][ln];
    accv = __builtin_amdgcn_wmma_f32_16x16x32_f16(false, a, false, bf,
                                                  (short)0, accv, false, false);
    __syncthreads();   // before next slab overwrites sX
  }
#pragma unroll
  for (int r = 0; r < 8; ++r) {
    int m = m0 + r + (half << 3);   // D layout: lanes 0-15 -> M=r, 16-31 -> M=r+8
    int n = n0 + ln;
    long yi = ((long)b * O + m) * Ntok + n;
    float v = accv[r] + bias[m];
    if (acc) v += Y[yi];
    if (relu) v = fmaxf(v, 0.f);
    if (resid) v += resid[((long)b * O + m) * (Ntok / rrep) + n / rrep];
    Y[yi] = v;
  }
}

// ---------------------------------------------------------------------------
// Fused attention block: one workgroup (8 wave32) per (b, n).
//   gather -> pos-MLP (VALU) -> s,val in LDS -> WMMA h=relu(bn(Wa1*s))
//   -> WMMA logits (both transposed-conv rows) -> softmax(K) -> aggregate.
// ---------------------------------------------------------------------------
__global__ __launch_bounds__(256) void attn_kernel(
    const float* __restrict__ pos,
    const float* __restrict__ kf, const float* __restrict__ qf,
    const float* __restrict__ vf, const float* __restrict__ uf,
    const int* __restrict__ idx,
    const float* __restrict__ Wp1, const float* __restrict__ bp1,
    const float* __restrict__ gp1, const float* __restrict__ bep1,
    const float* __restrict__ Wp2, const float* __restrict__ bp2,
    const float* __restrict__ Wa1, const float* __restrict__ ba1,
    const float* __restrict__ ga1, const float* __restrict__ bea1,
    const float* __restrict__ Wt, const float* __restrict__ bt,
    float* __restrict__ agg) {
  __shared__ int      sIdx[KNN];
  __shared__ float    sPR[3][KNN];
  __shared__ float    sH1[PHH][KNN];
  __shared__ _Float16 sS[DD][32];        // qk_rel + pos_emb + u_rel, K padded to 32
  __shared__ float    sVal[DD][KNN];
  __shared__ _Float16 sH[AHH][32];
  __shared__ float    sAtt[DD * UPF][32];

  const int tid = threadIdx.x;
  const int n = blockIdx.x;
  const int b = blockIdx.y;
  const float invs = 0.99999500003749973f;  // 1/sqrt(1+1e-5)

  if (tid == 0) __builtin_amdgcn_s_wait_tensorcnt((short)0);

  if (tid < KNN) sIdx[tid] = idx[((long)b * Npt + n) * KNN + tid];
  __syncthreads();
  if (tid < 3 * KNN) {
    int c = tid / KNN, k = tid % KNN;
    const float* pb = pos + ((long)b * 3 + c) * Npt;
    sPR[c][k] = pb[n] - pb[sIdx[k]];
  }
  __syncthreads();
  // pos_mlp hidden (3 -> 64, BN, ReLU): VALU, shapes too ragged for WMMA
  for (int it = tid; it < PHH * KNN; it += 256) {
    int ph = it / KNN, k = it % KNN;
    float a = Wp1[ph * 3 + 0] * sPR[0][k] + Wp1[ph * 3 + 1] * sPR[1][k] +
              Wp1[ph * 3 + 2] * sPR[2][k] + bp1[ph];
    a = gp1[ph] * a * invs + bep1[ph];
    sH1[ph][k] = fmaxf(a, 0.f);
  }
  __syncthreads();
  // pos_emb (64x64 per column), build s and val
  for (int it = tid; it < DD * KNN; it += 256) {
    int d = it / KNN, k = it % KNN;
    float pe = bp2[d];
#pragma unroll 8
    for (int ph = 0; ph < PHH; ++ph) pe += Wp2[d * PHH + ph] * sH1[ph][k];
    int j = sIdx[k];
    long base = ((long)b * DD + d) * Npt;
    float urel = uf[base + n] - uf[base + j];
    float qk   = qf[base + n] - kf[base + j];
    sS[d][k]   = (_Float16)(qk + pe + urel);
    sVal[d][k] = vf[base + j] + pe + urel;
  }
  for (int it = tid; it < DD * (32 - KNN); it += 256) {
    int d = it / (32 - KNN), k = KNN + it % (32 - KNN);
    sS[d][k] = (_Float16)0.f;
  }
  __syncthreads();

  const int w = tid >> 5, lane = tid & 31;
  const int half = lane >> 4, ln = lane & 15;

  // Phase B: h = relu(bn(Wa1 @ s)) -- (256x64)x(64x32), 32 tiles / 8 waves
#pragma unroll
  for (int i = 0; i < 4; ++i) {
    int T = w * 4 + i;
    int m0 = (T >> 1) * 16, n0 = (T & 1) * 16;
    v8f accv = {0.f, 0.f, 0.f, 0.f, 0.f, 0.f, 0.f, 0.f};
#pragma unroll
    for (int c0 = 0; c0 < DD; c0 += 32) {
      v16h a = load_a_rowmajor(Wa1, DD, m0, c0, lane);
      v16h bf;
#pragma unroll
      for (int q = 0; q < 16; ++q) bf[q] = sS[c0 + frag_k(q, half)][n0 + ln];
      accv = __builtin_amdgcn_wmma_f32_16x16x32_f16(false, a, false, bf,
                                                    (short)0, accv, false, false);
    }
#pragma unroll
    for (int r = 0; r < 8; ++r) {
      int m = m0 + r + (half << 3);
      float hv = ga1[m] * (accv[r] + ba1[m]) * invs + bea1[m];
      sH[m][n0 + ln] = (_Float16)fmaxf(hv, 0.f);
    }
  }
  __syncthreads();

  // Phase C: logits att[c][k] = sum_a h[a][k] * Wt[a][c][r] for r=0,1
  // A'[c][a] = Wt[a*(D*UP) + c*UP + r]; (64x256)x(256x32), 16 tiles / 8 waves
#pragma unroll
  for (int i = 0; i < 2; ++i) {
    int T = w * 2 + i;
    int rr = T >> 3;
    int m0 = ((T & 7) >> 1) * 16, n0 = (T & 1) * 16;
    v8f accv = {0.f, 0.f, 0.f, 0.f, 0.f, 0.f, 0.f, 0.f};
#pragma unroll
    for (int c0 = 0; c0 < AHH; c0 += 32) {
      v16h a, bf;
#pragma unroll
      for (int q = 0; q < 16; ++q) {
        int kk = c0 + frag_k(q, half);
        a[q]  = (_Float16)Wt[(long)kk * (DD * UPF) + (m0 + ln) * UPF + rr];
        bf[q] = sH[kk][n0 + ln];
      }
      accv = __builtin_amdgcn_wmma_f32_16x16x32_f16(false, a, false, bf,
                                                    (short)0, accv, false, false);
    }
#pragma unroll
    for (int r = 0; r < 8; ++r) {
      int m = m0 + r + (half << 3);
      sAtt[rr * DD + m][n0 + ln] = accv[r] + bt[m];
    }
  }
  __syncthreads();

  // Phase D: softmax over K (20 valid cols) + weighted aggregation
  if (tid < DD * UPF) {
    int rr = tid >> 6, c = tid & 63;
    float mx = -3.4e38f;
#pragma unroll
    for (int k = 0; k < KNN; ++k) mx = fmaxf(mx, sAtt[tid][k]);
    float s = 0.f, a = 0.f;
#pragma unroll
    for (int k = 0; k < KNN; ++k) {
      float e = __expf(sAtt[tid][k] - mx);
      s += e;
      a += e * sVal[c][k];
    }
    agg[((long)b * DD + c) * (Npt * UPF) + n * UPF + rr] = a / s;
  }
}

// ---------------------------------------------------------------------------
extern "C" void kernel_launch(void* const* d_in, const int* in_sizes, int n_in,
                              void* d_out, int out_size, void* d_ws, size_t ws_size,
                              hipStream_t stream) {
  (void)in_sizes; (void)n_in; (void)out_size; (void)ws_size;
  const float* pos  = (const float*)d_in[0];
  const float* keyf = (const float*)d_in[1];
  const float* qryf = (const float*)d_in[2];
  const float* upf  = (const float*)d_in[3];
  const float* Wv1  = (const float*)d_in[4];  const float* bv1  = (const float*)d_in[5];
  const float* Wv2  = (const float*)d_in[6];  const float* bv2  = (const float*)d_in[7];
  const float* Wvs  = (const float*)d_in[8];  const float* bvs  = (const float*)d_in[9];
  const float* Wk   = (const float*)d_in[10]; const float* bk   = (const float*)d_in[11];
  const float* Wq   = (const float*)d_in[12]; const float* bq   = (const float*)d_in[13];
  const float* Wvv  = (const float*)d_in[14]; const float* bvv  = (const float*)d_in[15];
  const float* Wu   = (const float*)d_in[16]; const float* bu   = (const float*)d_in[17];
  const float* Wp1  = (const float*)d_in[18]; const float* bp1  = (const float*)d_in[19];
  const float* gp1  = (const float*)d_in[20]; const float* bep1 = (const float*)d_in[21];
  const float* Wp2  = (const float*)d_in[22]; const float* bp2  = (const float*)d_in[23];
  const float* Wa1  = (const float*)d_in[24]; const float* ba1  = (const float*)d_in[25];
  const float* ga1  = (const float*)d_in[26]; const float* bea1 = (const float*)d_in[27];
  const float* Wt   = (const float*)d_in[28]; const float* bt   = (const float*)d_in[29];
  const float* Wend = (const float*)d_in[30]; const float* bend = (const float*)d_in[31];
  const float* Wres = (const float*)d_in[32]; const float* bres = (const float*)d_in[33];

  char* ws = (char*)d_ws;
  size_t off = 0;
  int*   idx    = (int*)(ws + off);   off += (size_t)Bn * Npt * KNN * sizeof(int);
  float* hidden = (float*)(ws + off); off += (size_t)Bn * CINC * Npt * sizeof(float);
  float* value  = (float*)(ws + off); off += (size_t)Bn * CINC * Npt * sizeof(float);
  float* kfb    = (float*)(ws + off); off += (size_t)Bn * DD * Npt * sizeof(float);
  float* qfb    = (float*)(ws + off); off += (size_t)Bn * DD * Npt * sizeof(float);
  float* vfb    = (float*)(ws + off); off += (size_t)Bn * DD * Npt * sizeof(float);
  float* ufb    = (float*)(ws + off); off += (size_t)Bn * DD * Npt * sizeof(float);
  float* resb   = (float*)(ws + off); off += (size_t)Bn * COUTC * Npt * sizeof(float);
  float* aggb   = (float*)(ws + off); off += (size_t)Bn * DD * Npt * UPF * sizeof(float);

  knn_kernel<<<dim3((Bn * Npt + 255) / 256), dim3(256), 0, stream>>>(pos, idx);

  dim3 t128(128);
  // mlp_v: hidden = relu(Wv1 * concat + bv1)
  gemm16<<<dim3(Npt / 16, CINC / 64, Bn), t128, 0, stream>>>(
      keyf, qryf, CINC, Wv1, bv1, nullptr, 1, hidden, CINC, 2 * CINC, Npt, 1, 0);
  // value = Wv2*hidden + bv2 ; value += Wvs*concat + bvs
  gemm16<<<dim3(Npt / 16, CINC / 64, Bn), t128, 0, stream>>>(
      hidden, hidden, CINC, Wv2, bv2, nullptr, 1, value, CINC, CINC, Npt, 0, 0);
  gemm16<<<dim3(Npt / 16, CINC / 64, Bn), t128, 0, stream>>>(
      keyf, qryf, CINC, Wvs, bvs, nullptr, 1, value, CINC, 2 * CINC, Npt, 0, 1);
  // projections CIN -> D
  gemm16<<<dim3(Npt / 16, DD / 64, Bn), t128, 0, stream>>>(
      keyf, keyf, CINC, Wk, bk, nullptr, 1, kfb, DD, CINC, Npt, 0, 0);
  gemm16<<<dim3(Npt / 16, DD / 64, Bn), t128, 0, stream>>>(
      qryf, qryf, CINC, Wq, bq, nullptr, 1, qfb, DD, CINC, Npt, 0, 0);
  gemm16<<<dim3(Npt / 16, DD / 64, Bn), t128, 0, stream>>>(
      value, value, CINC, Wvv, bvv, nullptr, 1, vfb, DD, CINC, Npt, 0, 0);
  gemm16<<<dim3(Npt / 16, DD / 64, Bn), t128, 0, stream>>>(
      upf, upf, CINC, Wu, bu, nullptr, 1, ufb, DD, CINC, Npt, 0, 0);
  // residual path: res = Wres * value + bres
  gemm16<<<dim3(Npt / 16, COUTC / 64, Bn), t128, 0, stream>>>(
      value, value, CINC, Wres, bres, nullptr, 1, resb, COUTC, CINC, Npt, 0, 0);

  // fused attention -> agg (B, D, N*UP)
  attn_kernel<<<dim3(Npt, Bn), dim3(256), 0, stream>>>(
      pos, kfb, qfb, vfb, ufb, idx, Wp1, bp1, gp1, bep1, Wp2, bp2,
      Wa1, ba1, ga1, bea1, Wt, bt, aggb);

  // conv_end + repeated residual -> out (B, COUT, N*UP)
  gemm16<<<dim3(Npt * UPF / 16, COUTC / 64, Bn), t128, 0, stream>>>(
      aggb, aggb, DD, Wend, bend, resb, UPF, (float*)d_out, COUTC, DD,
      Npt * UPF, 0, 0);
}
